// CosformerAttention_75806172774455
// MI455X (gfx1250) — compile-verified, matrix operand order checked by reading, not run on hardware
//
#include <hip/hip_runtime.h>
#include <hip/hip_bf16.h>

typedef __attribute__((ext_vector_type(16))) __bf16 v16bf;
typedef __attribute__((ext_vector_type(8)))  float  v8f;

constexpr int EDIM  = 768;
constexpr int LQ    = 2304;   // 48*48
constexpr int BATCH = 8;
constexpr int MTOT  = LQ * BATCH;   // 18432
constexpr int NHEAD = 96;           // B * 12
constexpr int BM = 128, BN = 128, BK = 32;
constexpr int BKP = BK + 8;         // LDS row pad (80B stride, 16B aligned)

// angle scale: (pi/2) / L
#define ANG_SCALE 0.00068165307074934f

// ---- CDNA5 async global->LDS path (guarded; fallback = sync load + ds_store)
#if defined(__has_builtin)
#if __has_builtin(__builtin_amdgcn_global_load_async_to_lds_b128) && \
    __has_builtin(__builtin_amdgcn_s_wait_asynccnt)
#define USE_ASYNC_LDS 1
#endif
#endif

#if defined(USE_ASYNC_LDS)
typedef int v4i_ __attribute__((ext_vector_type(4)));
typedef __attribute__((address_space(1))) v4i_* g128p;
typedef __attribute__((address_space(3))) v4i_* l128p;
__device__ inline void async_b128(const void* g, void* l) {
    __builtin_amdgcn_global_load_async_to_lds_b128((g128p)g, (l128p)l, 0, 0);
}
__device__ inline void wait_async0() { __builtin_amdgcn_s_wait_asynccnt(0); }
#else
__device__ inline void wait_async0() {}
#endif

__device__ inline v16bf load16bf(const __bf16* p) {
    union { v16bf v; uint4 u[2]; } r;
    r.u[0] = *(const uint4*)p;
    r.u[1] = *(const uint4*)(p + 8);
    return r.v;
}

// ---------------------------------------------------------------------------
// Pack query (B, E, L) f32 -> x (M = l*B+b, E) bf16 via LDS tile transpose
// ---------------------------------------------------------------------------
__global__ __launch_bounds__(256)
void pack_x_kernel(const float* __restrict__ query, __bf16* __restrict__ xb) {
    __shared__ float tile[32][33];
    const int l0 = blockIdx.x * 32;
    const int e0 = blockIdx.y * 32;
    const int b  = blockIdx.z;
    const int tx = threadIdx.x;      // 0..31 (along L on read)
    const int ty = threadIdx.y;      // 0..7
    #pragma unroll
    for (int i = 0; i < 32; i += 8) {
        const int e = e0 + ty + i;
        tile[ty + i][tx] = query[((size_t)b * EDIM + e) * LQ + l0 + tx];
    }
    __syncthreads();
    #pragma unroll
    for (int i = 0; i < 32; i += 8) {
        const int l = l0 + ty + i;
        const int m = l * BATCH + b;
        xb[(size_t)m * EDIM + e0 + tx] = (__bf16)tile[tx][ty + i];
    }
}

// ---------------------------------------------------------------------------
// Pack Wq|Wk|Wv -> fused (2304 x 768) bf16, Wo -> (768 x 768) bf16
// ---------------------------------------------------------------------------
__global__ __launch_bounds__(256)
void pack_w_kernel(const float* __restrict__ Wq, const float* __restrict__ Wk,
                   const float* __restrict__ Wv, const float* __restrict__ Wo,
                   __bf16* __restrict__ Wqkv_b, __bf16* __restrict__ Wo_b) {
    const int idx = blockIdx.x * 256 + threadIdx.x;
    const int EE = EDIM * EDIM;
    if (idx < 3 * EE) {
        float v;
        if (idx < EE)            v = Wq[idx];
        else if (idx < 2 * EE)   v = Wk[idx - EE];
        else                     v = Wv[idx - 2 * EE];
        Wqkv_b[idx] = (__bf16)v;
    } else if (idx < 4 * EE) {
        Wo_b[idx - 3 * EE] = (__bf16)Wo[idx - 3 * EE];
    }
}

// ---------------------------------------------------------------------------
// bf16 WMMA GEMM: Y = A(MxK) * W(NxK)^T.  128x128x32 tiles, double-buffered
// async global->LDS staging, 8 waves, each wave -> 32x64 via 2x4
// v_wmma_f32_16x16x32_bf16 accumulators.
// MODE 0: fused QKV epilogue (bias, relu, sin/cos head-split scatter)
// MODE 1: output projection epilogue (bias -> f32 d_out)
// ---------------------------------------------------------------------------
template <int MODE>
__global__ __launch_bounds__(256)
void gemm_bf16_kernel(const __bf16* __restrict__ A, const __bf16* __restrict__ W,
                      const float* __restrict__ bias0, const float* __restrict__ bias1,
                      const float* __restrict__ bias2,
                      __bf16* __restrict__ qhat, __bf16* __restrict__ khat,
                      float* __restrict__ vbuf, float* __restrict__ outp) {
    __shared__ __bf16 As[2][BM][BKP];
    __shared__ __bf16 Bs[2][BN][BKP];

    const int tid  = threadIdx.x;
    const int t31  = tid & 31;
    const int wave = tid >> 5;       // 0..7
    const int wm   = wave >> 1;      // 0..3 (M)
    const int wn   = wave & 1;       // 0..1 (N)
    const int m0   = blockIdx.y * BM;
    const int n0   = blockIdx.x * BN;

    v8f c[2][4];
    #pragma unroll
    for (int i = 0; i < 2; i++)
        #pragma unroll
        for (int j = 0; j < 4; j++)
            c[i][j] = v8f{};

    const int lrow = tid >> 1;            // 0..127
    const int lks  = (tid & 1) * 16;      // 0 | 16

    const __bf16* ga = A + (size_t)(m0 + lrow) * EDIM + lks;
    const __bf16* gb = W + (size_t)(n0 + lrow) * EDIM + lks;

    auto stage = [&](int buf, int kt) {
#if defined(USE_ASYNC_LDS)
        async_b128(ga + kt,     &As[buf][lrow][lks]);
        async_b128(ga + kt + 8, &As[buf][lrow][lks + 8]);
        async_b128(gb + kt,     &Bs[buf][lrow][lks]);
        async_b128(gb + kt + 8, &Bs[buf][lrow][lks + 8]);
#else
        uint4 a0 = *(const uint4*)(ga + kt);
        uint4 a1 = *(const uint4*)(ga + kt + 8);
        uint4 b0 = *(const uint4*)(gb + kt);
        uint4 b1 = *(const uint4*)(gb + kt + 8);
        if (kt + BK < EDIM) {            // prefetch next K tile -> global_prefetch_b8
            __builtin_prefetch(ga + kt + BK, 0, 3);
            __builtin_prefetch(gb + kt + BK, 0, 3);
        }
        *(uint4*)&As[buf][lrow][lks]     = a0;
        *(uint4*)&As[buf][lrow][lks + 8] = a1;
        *(uint4*)&Bs[buf][lrow][lks]     = b0;
        *(uint4*)&Bs[buf][lrow][lks + 8] = b1;
#endif
    };

    stage(0, 0);
    int buf = 0;
    const int khalf = (t31 >> 4) * 16;
    const int lr    = t31 & 15;

    for (int kt = 0; kt < EDIM; kt += BK) {
        wait_async0();        // this wave's staged tile is in LDS
        __syncthreads();      // everyone's tile is visible
        if (kt + BK < EDIM) stage(buf ^ 1, kt + BK);   // overlap next stage

        v16bf afrag[2], bfrag[4];
        #pragma unroll
        for (int i = 0; i < 2; i++)
            afrag[i] = load16bf(&As[buf][wm * 32 + i * 16 + lr][khalf]);
        #pragma unroll
        for (int j = 0; j < 4; j++)
            bfrag[j] = load16bf(&Bs[buf][wn * 64 + j * 16 + lr][khalf]);
        #pragma unroll
        for (int i = 0; i < 2; i++)
            #pragma unroll
            for (int j = 0; j < 4; j++)
                c[i][j] = __builtin_amdgcn_wmma_f32_16x16x32_bf16(
                    false, afrag[i], false, bfrag[j], (short)0, c[i][j], false, false);
        __syncthreads();      // done reading buf before it is restaged
        buf ^= 1;
    }

    // epilogue: accumulator VGPR r holds M = mbase + r, N = n0 base + lane&15.
    // mbase is a multiple of 8 => l = m>>3 is constant per accumulator,
    // b = m&7 = r. One sincos per i instead of per element.
    const int hi = (t31 >> 4) * 8;
    #pragma unroll
    for (int i = 0; i < 2; i++) {
        const int mbase = m0 + wm * 32 + i * 16 + hi;
        const int l = mbase >> 3;
        float s = 0.f, co = 0.f;
        if (MODE == 0) {
            const float ang = (float)(l + 1) * ANG_SCALE;
            __sincosf(ang, &s, &co);
        }
        #pragma unroll
        for (int j = 0; j < 4; j++) {
            const int n = n0 + wn * 64 + j * 16 + lr;
            #pragma unroll
            for (int r = 0; r < 8; r++) {
                const float y = c[i][j][r];
                if (MODE == 1) {
                    outp[(size_t)(mbase + r) * EDIM + n] = y + bias0[n];
                } else if (n < EDIM) {                 // Q: relu + sin/cos split
                    const float t = fmaxf(y + bias0[n], 0.f);
                    const int hd = n >> 6, dd = n & 63;
                    const size_t base = ((size_t)(r * 12 + hd) * LQ + l) * 128 + dd;
                    qhat[base]      = (__bf16)(t * s);
                    qhat[base + 64] = (__bf16)(t * co);
                } else if (n < 2 * EDIM) {             // K
                    const int e = n - EDIM;
                    const float t = fmaxf(y + bias1[e], 0.f);
                    const int hd = e >> 6, dd = e & 63;
                    const size_t base = ((size_t)(r * 12 + hd) * LQ + l) * 128 + dd;
                    khat[base]      = (__bf16)(t * s);
                    khat[base + 64] = (__bf16)(t * co);
                } else {                               // V (f32)
                    const int e = n - 2 * EDIM;
                    const int hd = e >> 6, dd = e & 63;
                    vbuf[((size_t)(r * 12 + hd) * LQ + l) * 64 + dd] = y + bias2[e];
                }
            }
        }
    }
}

// ---------------------------------------------------------------------------
// Per head: kv[128][64] = sum_l khat[l]^T outer v[l];  ksum[128] = sum_l khat[l]
// ---------------------------------------------------------------------------
__global__ __launch_bounds__(256)
void stage2a_kernel(const __bf16* __restrict__ khat, const float* __restrict__ vbuf,
                    float* __restrict__ kv, float* __restrict__ ksum) {
    const int n = blockIdx.x;
    const int t = threadIdx.x;
    const int dd2 = t >> 1;          // 0..127
    const int mb  = (t & 1) * 32;    // 0 | 32
    __shared__ __bf16 kS[64][128];
    __shared__ float  vS[64][68];
    float acc[32];
    #pragma unroll
    for (int j = 0; j < 32; j++) acc[j] = 0.f;
    float ks = 0.f;

    const __bf16* kbase = khat + (size_t)n * LQ * 128;
    const float*  vbase = vbuf + (size_t)n * LQ * 64;

    for (int l0 = 0; l0 < LQ; l0 += 64) {
        for (int i = t; i < 64 * 128 / 8; i += 256) {   // 4 iters: 8 bf16 each
            const int row = i >> 4, col = (i & 15) * 8;
            const __bf16* src = kbase + (size_t)(l0 + row) * 128 + col;
#if defined(USE_ASYNC_LDS)
            async_b128(src, &kS[row][col]);
#else
            *(uint4*)&kS[row][col] = *(const uint4*)src;
#endif
        }
        for (int i = t; i < 64 * 64 / 4; i += 256) {    // 4 iters: float4 each
            const int row = i >> 4, col = (i & 15) * 4;
            const float* src = vbase + (size_t)(l0 + row) * 64 + col;
#if defined(USE_ASYNC_LDS)
            async_b128(src, &vS[row][col]);
#else
            *(float4*)&vS[row][col] = *(const float4*)src;
#endif
        }
        wait_async0();
        __syncthreads();
        for (int lc = 0; lc < 64; lc++) {
            const float kval = (float)kS[lc][dd2];
            ks += kval;
            const float* vr = &vS[lc][mb];
            #pragma unroll
            for (int j = 0; j < 32; j++) acc[j] = fmaf(kval, vr[j], acc[j]);
        }
        __syncthreads();
    }
    float* kvout = kv + (size_t)n * (128 * 64) + dd2 * 64 + mb;
    #pragma unroll
    for (int j = 0; j < 32; j++) kvout[j] = acc[j];
    if ((t & 1) == 0) ksum[n * 128 + dd2] = ks;
}

// ---------------------------------------------------------------------------
// out = (qhat . kv) / max(qhat . ksum, eps); attn_bf = bf16(x + out)
// 256 threads = 4 rows of (n,l), 64 threads per row (one per output dim)
// ---------------------------------------------------------------------------
__global__ __launch_bounds__(256)
void stage2b_kernel(const __bf16* __restrict__ qhat, const float* __restrict__ kv,
                    const float* __restrict__ ksum, const float* __restrict__ query,
                    __bf16* __restrict__ attn_bf) {
    const int sub = threadIdx.x >> 6;
    const int mm  = threadIdx.x & 63;
    const int rid = blockIdx.x * 4 + sub;       // n*LQ + l
    const int n = rid / LQ;
    const int l = rid - n * LQ;
    __shared__ float qs[4][128];
    const __bf16* qr = qhat + (size_t)rid * 128;
    qs[sub][mm]      = (float)qr[mm];
    qs[sub][mm + 64] = (float)qr[mm + 64];
    __syncthreads();
    const float* kvp = kv + (size_t)n * (128 * 64) + mm;
    const float* ksp = ksum + n * 128;
    float acc = 0.f, qdot = 0.f;
    #pragma unroll 8
    for (int dd2 = 0; dd2 < 128; dd2++) {
        const float qv = qs[sub][dd2];
        acc  = fmaf(qv, kvp[(size_t)dd2 * 64], acc);
        qdot = fmaf(qv, ksp[dd2], qdot);
    }
    const float z = 1.f / fmaxf(qdot, 1e-6f);
    const int b = n / 12, hd = n - b * 12;
    const int e = hd * 64 + mm;
    const int m = l * BATCH + b;
    const float x = query[((size_t)b * EDIM + e) * LQ + l];
    attn_bf[(size_t)m * EDIM + e] = (__bf16)(x + acc * z);
}

// ---------------------------------------------------------------------------
extern "C" void kernel_launch(void* const* d_in, const int* in_sizes, int n_in,
                              void* d_out, int out_size, void* d_ws, size_t ws_size,
                              hipStream_t stream) {
    const float* query = (const float*)d_in[0];
    const float* Wq = (const float*)d_in[1];
    const float* bq = (const float*)d_in[2];
    const float* Wk = (const float*)d_in[3];
    const float* bk = (const float*)d_in[4];
    const float* Wv = (const float*)d_in[5];
    const float* bv = (const float*)d_in[6];
    const float* Wo = (const float*)d_in[7];
    const float* bo = (const float*)d_in[8];
    float* out = (float*)d_out;

    char* ws = (char*)d_ws;
    size_t off = 0;
    auto alloc = [&](size_t bytes) -> void* {
        void* p = ws + off;
        off = (off + bytes + 255) & ~(size_t)255;
        return p;
    };
    __bf16* xb      = (__bf16*)alloc((size_t)MTOT * EDIM * 2);        // 28.3 MB
    __bf16* Wqkv_b  = (__bf16*)alloc((size_t)3 * EDIM * EDIM * 2);    //  3.5 MB
    __bf16* Wo_b    = (__bf16*)alloc((size_t)EDIM * EDIM * 2);        //  1.2 MB
    __bf16* qhat    = (__bf16*)alloc((size_t)NHEAD * LQ * 128 * 2);   // 56.6 MB
    __bf16* khat    = (__bf16*)alloc((size_t)NHEAD * LQ * 128 * 2);   // 56.6 MB
    float*  vbuf    = (float*) alloc((size_t)NHEAD * LQ * 64 * 4);    // 56.6 MB
    float*  kv      = (float*) alloc((size_t)NHEAD * 128 * 64 * 4);   //  3.1 MB
    float*  ksum    = (float*) alloc((size_t)NHEAD * 128 * 4);        // 49 KB
    __bf16* attn_bf = (__bf16*)alloc((size_t)MTOT * EDIM * 2);        // 28.3 MB

    // 1) pack x (transpose) and weights to bf16
    pack_x_kernel<<<dim3(LQ / 32, EDIM / 32, BATCH), dim3(32, 8), 0, stream>>>(query, xb);
    pack_w_kernel<<<(4 * EDIM * EDIM + 255) / 256, 256, 0, stream>>>(Wq, Wk, Wv, Wo, Wqkv_b, Wo_b);

    // 2) fused QKV projection GEMM (M=18432, N=2304, K=768) + epilogue
    gemm_bf16_kernel<0><<<dim3(3 * EDIM / BN, MTOT / BM), 256, 0, stream>>>(
        xb, Wqkv_b, bq, bk, bv, qhat, khat, vbuf, nullptr);

    // 3) linear-attention rank-128 state per head
    stage2a_kernel<<<NHEAD, 256, 0, stream>>>(khat, vbuf, kv, ksum);

    // 4) apply state + normalizer + residual
    stage2b_kernel<<<NHEAD * LQ / 4, 256, 0, stream>>>(qhat, kv, ksum, query, attn_bf);

    // 5) output projection GEMM (M=18432, N=768, K=768)
    gemm_bf16_kernel<1><<<dim3(EDIM / BN, MTOT / BM), 256, 0, stream>>>(
        attn_bf, Wo_b, bo, nullptr, nullptr, nullptr, nullptr, nullptr, out);

    (void)in_sizes; (void)n_in; (void)out_size; (void)ws_size;
}
